// ViTMicro_25288767438871
// MI455X (gfx1250) — matrix operand order repaired and checked by reading, not compiled, not run on hardware
//
#include <hip/hip_runtime.h>

typedef __attribute__((ext_vector_type(16))) _Float16 v16h;
typedef __attribute__((ext_vector_type(8)))  float    v8f;

#define DIM   32
#define NTOK  4
#define BATCH 262144
#define ROWS  (BATCH*NTOK)
#define TILES (ROWS/16)
#define NT    18            // number of 32x16 f16 weight tiles
#define STR   68            // LDS staging row stride (floats)
#define SCALE 0.17677669529663687f  // 1/sqrt(32)

// f32 param-table offsets (floats)
#define P_PB  0
#define P_L1G 128
#define P_L1B 160
#define P_BQ  192
#define P_BK  224
#define P_BV  256
#define P_BO  288
#define P_L2G 320
#define P_L2B 352
#define P_BF1 384
#define P_BF2 448
#define P_TOT 480

#define LDS_FENCE() __asm__ volatile("s_wait_dscnt 0" ::: "memory")

struct VitPtrs {
  const float *w_pe,*b_pe,*pos,*ln1_g,*ln1_b,*wq,*bq,*wk,*bk,*wv,*bv,
              *wo,*bo,*ln2_g,*ln2_b,*w1,*bf1,*w2,*bf2;
};

// ---------------- pre-kernel: repack weights to WMMA B-fragment f16 blobs ----
// Blob layout: tile t, lane l, j=0..15 halfs; value = B[kOff + 8*(l>>4) + j + 8*(j>>3)][nOff + (l&15)]
__global__ __launch_bounds__(256) void vit_prep(VitPtrs P, _Float16* wsW, float* wsP){
  const int tid = threadIdx.x;
  for (int idx = tid; idx < NT*512; idx += 256){
    const int tile = idx >> 9, lane = (idx >> 4) & 31, j = idx & 15;
    const int n = lane & 15, hi = lane >> 4;
    const float* W; int ld = 32, kOff = 0, nOff;
    if (tile < 10){
      const float* Ws[5] = {P.w_pe, P.wq, P.wk, P.wv, P.wo};
      W = Ws[tile >> 1]; nOff = 16*(tile & 1);
    } else if (tile < 14){
      W = P.w1; ld = 64; nOff = 16*(tile-10);
    } else {
      W = P.w2; kOff = 32*((tile-14)>>1); nOff = 16*((tile-14)&1);
    }
    const int k = kOff + 8*hi + j + 8*(j>>3);
    wsW[idx] = (_Float16)W[k*ld + nOff + n];
  }
  for (int d = tid; d < P_TOT; d += 256){
    float v;
    if      (d < 128) v = P.b_pe[d & 31] + P.pos[d];   // fused patch bias + pos
    else if (d < 160) v = P.ln1_g[d-128];
    else if (d < 192) v = P.ln1_b[d-160];
    else if (d < 224) v = P.bq[d-192];
    else if (d < 256) v = P.bk[d-224];
    else if (d < 288) v = P.bv[d-256];
    else if (d < 320) v = P.bo[d-288];
    else if (d < 352) v = P.ln2_g[d-320];
    else if (d < 384) v = P.ln2_b[d-352];
    else if (d < 448) v = P.bf1[d-384];
    else              v = P.bf2[d-448];
    wsP[d] = v;
  }
}

// ---------------- device helpers -------------------------------------------
__device__ __forceinline__ v8f wmma_f16(v16h a, v16h b, v8f c){
  return __builtin_amdgcn_wmma_f32_16x16x32_f16(false, a, false, b, (short)0, c,
                                                false, false);
}

// DPP rotate-tree add (stays in VALU; no LDS/bpermute traffic)
template<int CTRL>
__device__ __forceinline__ float dpp_add(float x){
  const int r = __builtin_amdgcn_update_dpp(0, __float_as_int(x), CTRL, 0xF, 0xF, false);
  return x + __int_as_float(r);
}
// sum over each 16-lane half (row_ror:8,4,2,1)
__device__ __forceinline__ float red16(float x){
  x = dpp_add<0x128>(x);
  x = dpp_add<0x124>(x);
  x = dpp_add<0x122>(x);
  x = dpp_add<0x121>(x);
  return x;
}

// write one 16x32 C-layout fragment pair to per-wave staging (row-major f32)
__device__ __forceinline__ void stageC(float* st, const v8f& c0, const v8f& c1,
                                       int n, int hi){
  LDS_FENCE();
  #pragma unroll
  for (int i = 0; i < 8; ++i){
    st[(i + 8*hi)*STR + n]      = c0[i];
    st[(i + 8*hi)*STR + 16 + n] = c1[i];
  }
  LDS_FENCE();
}

// A-fragment (16xK window starting at col kb): lane<16 K kb+{0..7,16..23}, lane>=16 +8
__device__ __forceinline__ v16h ldsA(const float* st, int lane, int kb){
  const int m = lane & 15, hi = lane >> 4;
  const float4* r0 = (const float4*)(st + m*STR + kb + 8*hi);
  const float4* r1 = (const float4*)(st + m*STR + kb + 16 + 8*hi);
  const float4 a0 = r0[0], a1 = r0[1], b0 = r1[0], b1 = r1[1];
  v16h a;
  a[0]=(_Float16)a0.x; a[1]=(_Float16)a0.y; a[2]=(_Float16)a0.z; a[3]=(_Float16)a0.w;
  a[4]=(_Float16)a1.x; a[5]=(_Float16)a1.y; a[6]=(_Float16)a1.z; a[7]=(_Float16)a1.w;
  a[8]=(_Float16)b0.x; a[9]=(_Float16)b0.y; a[10]=(_Float16)b0.z; a[11]=(_Float16)b0.w;
  a[12]=(_Float16)b1.x; a[13]=(_Float16)b1.y; a[14]=(_Float16)b1.z; a[15]=(_Float16)b1.w;
  return a;
}

// B-fragment of (staged matrix)^T (32x16): B[k][n] = staged[n][k]
__device__ __forceinline__ v16h ldsBT(const float* st, int lane){
  const int nn = lane & 15, hi = lane >> 4;
  const float4* r = (const float4*)(st + nn*STR + 8*hi);
  const float4* r2 = (const float4*)(st + nn*STR + 16 + 8*hi);
  const float4 a0 = r[0], a1 = r[1], a2 = r2[0], a3 = r2[1];
  v16h b;
  b[0]=(_Float16)a0.x; b[1]=(_Float16)a0.y; b[2]=(_Float16)a0.z; b[3]=(_Float16)a0.w;
  b[4]=(_Float16)a1.x; b[5]=(_Float16)a1.y; b[6]=(_Float16)a1.z; b[7]=(_Float16)a1.w;
  b[8]=(_Float16)a2.x; b[9]=(_Float16)a2.y; b[10]=(_Float16)a2.z; b[11]=(_Float16)a2.w;
  b[12]=(_Float16)a3.x; b[13]=(_Float16)a3.y; b[14]=(_Float16)a3.z; b[15]=(_Float16)a3.w;
  return b;
}

__device__ __forceinline__ float fast_tanh(float u){
#if __has_builtin(__builtin_amdgcn_tanhf)
  return __builtin_amdgcn_tanhf(u);          // v_tanh_f32 (gfx1250 TRANS op)
#else
  const float au = fabsf(u);
  const float e  = __expf(-2.0f*au);
  float th = (1.0f - e) * __builtin_amdgcn_rcpf(1.0f + e);
  return copysignf(th, u);
#endif
}

__device__ __forceinline__ float gelu_tanh(float z){
  const float u = 0.7978845608028654f * (z + 0.044715f * z*z*z);
  return 0.5f * z * (1.0f + fast_tanh(u));
}

// ---------------- main fused kernel -----------------------------------------
__global__ __launch_bounds__(256) void vit_main(const float* __restrict__ xin,
                                                float* __restrict__ out,
                                                const _Float16* __restrict__ wsW,
                                                const float* __restrict__ wsP){
  __shared__ __align__(32) _Float16 sW[NT*512];   // 18 KB weight fragments
  __shared__ float sP[P_TOT];                     // biases / LN params
  __shared__ __align__(16) float sT[8][16*STR];   // per-wave transpose staging

  const int tid = threadIdx.x;
  {
    const uint4* s = (const uint4*)wsW; uint4* d = (uint4*)sW;
    for (int i = tid; i < (NT*512*2)/16; i += 256) d[i] = s[i];
    for (int i = tid; i < P_TOT; i += 256) sP[i] = wsP[i];
  }
  __syncthreads();

  const int lane = tid & 31, wave = tid >> 5;
  const int n = lane & 15, hi = lane >> 4;
  float* st = sT[wave];

  auto WF = [&](int t) -> v16h { return *(const v16h*)&sW[t*512 + lane*16]; };

  // per-lane column-resident parameters (fragment columns n and n+16)
  float pb0[4], pb1[4];
  #pragma unroll
  for (int t = 0; t < 4; ++t){ pb0[t] = sP[P_PB + t*32 + n];
                               pb1[t] = sP[P_PB + t*32 + 16 + n]; }
  const float l1g0=sP[P_L1G+n], l1g1=sP[P_L1G+16+n];
  const float l1b0=sP[P_L1B+n], l1b1=sP[P_L1B+16+n];
  const float bq0 =sP[P_BQ+n],  bq1 =sP[P_BQ+16+n];
  const float bk0 =sP[P_BK+n],  bk1 =sP[P_BK+16+n];
  const float bv0 =sP[P_BV+n],  bv1 =sP[P_BV+16+n];
  const float bo0 =sP[P_BO+n],  bo1 =sP[P_BO+16+n];
  const float l2g0=sP[P_L2G+n], l2g1=sP[P_L2G+16+n];
  const float l2b0=sP[P_L2B+n], l2b1=sP[P_L2B+16+n];
  float bf1v[4];
  #pragma unroll
  for (int c = 0; c < 4; ++c) bf1v[c] = sP[P_BF1 + c*16 + n];
  const float bf20=sP[P_BF2+n], bf21=sP[P_BF2+16+n];

  const v8f zero = {};
  const int nw = gridDim.x * 8;

  for (int t = blockIdx.x*8 + wave; t < TILES; t += nw){
    const float* xb = xin + (size_t)t * (16*DIM);

    // ---- load input tile straight in A-fragment pattern ----
    v16h ax;
    {
      const float* r = xb + (lane & 15)*DIM + 8*hi;
      const float4 f0 = *(const float4*)(r);
      const float4 f1 = *(const float4*)(r + 4);
      const float4 f2 = *(const float4*)(r + 16);
      const float4 f3 = *(const float4*)(r + 20);
      ax[0]=(_Float16)f0.x; ax[1]=(_Float16)f0.y; ax[2]=(_Float16)f0.z; ax[3]=(_Float16)f0.w;
      ax[4]=(_Float16)f1.x; ax[5]=(_Float16)f1.y; ax[6]=(_Float16)f1.z; ax[7]=(_Float16)f1.w;
      ax[8]=(_Float16)f2.x; ax[9]=(_Float16)f2.y; ax[10]=(_Float16)f2.z; ax[11]=(_Float16)f2.w;
      ax[12]=(_Float16)f3.x; ax[13]=(_Float16)f3.y; ax[14]=(_Float16)f3.z; ax[15]=(_Float16)f3.w;
    }

    // ---- patch embed: X = x @ w_pe + (b_pe + pos) ----
    v8f x0 = wmma_f16(ax, WF(0), zero);
    v8f x1 = wmma_f16(ax, WF(1), zero);
    #pragma unroll
    for (int i = 0; i < 8; ++i){ x0[i] += pb0[i & 3]; x1[i] += pb1[i & 3]; }

    // ---- LN1: single-pass E[x^2]-mean^2, concurrent DPP reductions ----
    v8f y0, y1;
    #pragma unroll
    for (int i = 0; i < 8; ++i){
      const float s  = red16(x0[i] + x1[i]);
      const float s2 = red16(__builtin_fmaf(x0[i], x0[i], x1[i]*x1[i]));
      const float mean = s * (1.0f/32.0f);
      const float var  = __builtin_fmaf(s2, 1.0f/32.0f, -mean*mean);
      const float rs = __builtin_amdgcn_rsqf(var + 1e-5f);
      y0[i] = (x0[i]-mean)*(rs*l1g0) + l1b0;
      y1[i] = (x1[i]-mean)*(rs*l1g1) + l1b1;
    }

    // ---- QKV (xa -> A-fragment via LDS transpose) ----
    stageC(st, y0, y1, n, hi);
    const v16h axa = ldsA(st, lane, 0);
    v8f q0 = wmma_f16(axa, WF(2), zero), q1 = wmma_f16(axa, WF(3), zero);
    v8f k0 = wmma_f16(axa, WF(4), zero), k1 = wmma_f16(axa, WF(5), zero);
    v8f v0 = wmma_f16(axa, WF(6), zero), v1 = wmma_f16(axa, WF(7), zero);
    #pragma unroll
    for (int i = 0; i < 8; ++i){
      q0[i] += bq0; q1[i] += bq1; k0[i] += bk0; k1[i] += bk1;
      v0[i] += bv0; v1[i] += bv1;
    }

    // ---- transposed scores: scT = K @ Q^T  (scT[s][m] = score[m][s]) ----
    stageC(st, q0, q1, n, hi);
    const v16h bQ = ldsBT(st, lane);
    stageC(st, k0, k1, n, hi);
    const v16h aK = ldsA(st, lane, 0);
    const v8f scT = wmma_f16(aK, bQ, zero);

    // ---- in-lane softmax over the 4 same-item keys of column m=n ----
    // valid owner lanes: hi == (n>=8); VGPR base vb = n&4; garbage elsewhere unused
    float aw4[4];
    {
      const bool up = (n & 4) != 0;
      float xs0 = (up ? scT[4] : scT[0]) * SCALE;
      float xs1 = (up ? scT[5] : scT[1]) * SCALE;
      float xs2 = (up ? scT[6] : scT[2]) * SCALE;
      float xs3 = (up ? scT[7] : scT[3]) * SCALE;
      const float mx = fmaxf(fmaxf(xs0, xs1), fmaxf(xs2, xs3));
      const float e0 = __expf(xs0 - mx), e1 = __expf(xs1 - mx);
      const float e2 = __expf(xs2 - mx), e3 = __expf(xs3 - mx);
      const float rs = __builtin_amdgcn_rcpf(e0 + e1 + e2 + e3);
      aw4[0] = e0*rs; aw4[1] = e1*rs; aw4[2] = e2*rs; aw4[3] = e3*rs;
    }

    // ---- mixed = attn @ V : broadcast weights via tiny LDS table ----
    // every lane publishes its slot; readers hit only owner slots {0..7, 24..31}
    LDS_FENCE();
    #pragma unroll
    for (int tt = 0; tt < 4; ++tt) st[lane*4 + tt] = aw4[tt];
    LDS_FENCE();
    v8f m0, m1;
    #pragma unroll
    for (int i = 0; i < 8; ++i){
      const float4 w = *(const float4*)(st + (i + 24*hi)*4);  // broadcast read
      const int b = i & ~3;
      m0[i] = w.x*v0[b] + w.y*v0[b+1] + w.z*v0[b+2] + w.w*v0[b+3];
      m1[i] = w.x*v1[b] + w.y*v1[b+1] + w.z*v1[b+2] + w.w*v1[b+3];
    }

    // ---- out-proj, residual fused as WMMA C operand ----
    stageC(st, m0, m1, n, hi);
    const v16h am = ldsA(st, lane, 0);
    x0 = wmma_f16(am, WF(8), x0);
    x1 = wmma_f16(am, WF(9), x1);
    #pragma unroll
    for (int i = 0; i < 8; ++i){ x0[i] += bo0; x1[i] += bo1; }

    // ---- LN2 ----
    #pragma unroll
    for (int i = 0; i < 8; ++i){
      const float s  = red16(x0[i] + x1[i]);
      const float s2 = red16(__builtin_fmaf(x0[i], x0[i], x1[i]*x1[i]));
      const float mean = s * (1.0f/32.0f);
      const float var  = __builtin_fmaf(s2, 1.0f/32.0f, -mean*mean);
      const float rs = __builtin_amdgcn_rsqf(var + 1e-5f);
      y0[i] = (x0[i]-mean)*(rs*l2g0) + l2b0;
      y1[i] = (x1[i]-mean)*(rs*l2g1) + l2b1;
    }

    // ---- MLP up: h = gelu(xb @ w1 + bf1), 16x64 ----
    stageC(st, y0, y1, n, hi);
    const v16h axm = ldsA(st, lane, 0);
    v8f h[4];
    #pragma unroll
    for (int c = 0; c < 4; ++c){
      h[c] = wmma_f16(axm, WF(10 + c), zero);
      #pragma unroll
      for (int i = 0; i < 8; ++i) h[c][i] = gelu_tanh(h[c][i] + bf1v[c]);
    }

    // ---- MLP down: X += h @ w2 + bf2 (K=64 -> two chained WMMAs per N-block) ----
    LDS_FENCE();
    #pragma unroll
    for (int c = 0; c < 4; ++c)
      #pragma unroll
      for (int i = 0; i < 8; ++i)
        st[(i + 8*hi)*STR + c*16 + n] = h[c][i];
    LDS_FENCE();
    const v16h ah0 = ldsA(st, lane, 0);
    const v16h ah1 = ldsA(st, lane, 32);
    x0 = wmma_f16(ah0, WF(14), x0);
    x0 = wmma_f16(ah1, WF(16), x0);
    x1 = wmma_f16(ah0, WF(15), x1);
    x1 = wmma_f16(ah1, WF(17), x1);
    #pragma unroll
    for (int i = 0; i < 8; ++i){ x0[i] += bf20; x1[i] += bf21; }

    // ---- store result tile ----
    float* ob = out + (size_t)t * (16*DIM);
    #pragma unroll
    for (int i = 0; i < 8; ++i){
      ob[(i + 8*hi)*DIM + n]      = x0[i];
      ob[(i + 8*hi)*DIM + 16 + n] = x1[i];
    }
  }
}

// ---------------- launch ----------------------------------------------------
extern "C" void kernel_launch(void* const* d_in, const int* in_sizes, int n_in,
                              void* d_out, int out_size, void* d_ws, size_t ws_size,
                              hipStream_t stream) {
  (void)in_sizes; (void)n_in; (void)out_size; (void)ws_size;
  _Float16* wsW = (_Float16*)d_ws;                       // 18432 B of f16 B-fragments
  float*    wsP = (float*)((char*)d_ws + NT*512*2);      // 1920 B of f32 params

  VitPtrs P;
  P.w_pe =(const float*)d_in[1];  P.b_pe =(const float*)d_in[2];
  P.pos  =(const float*)d_in[3];
  P.ln1_g=(const float*)d_in[4];  P.ln1_b=(const float*)d_in[5];
  P.wq   =(const float*)d_in[6];  P.bq   =(const float*)d_in[7];
  P.wk   =(const float*)d_in[8];  P.bk   =(const float*)d_in[9];
  P.wv   =(const float*)d_in[10]; P.bv   =(const float*)d_in[11];
  P.wo   =(const float*)d_in[12]; P.bo   =(const float*)d_in[13];
  P.ln2_g=(const float*)d_in[14]; P.ln2_b=(const float*)d_in[15];
  P.w1   =(const float*)d_in[16]; P.bf1  =(const float*)d_in[17];
  P.w2   =(const float*)d_in[18]; P.bf2  =(const float*)d_in[19];

  vit_prep<<<1, 256, 0, stream>>>(P, wsW, wsP);
  vit_main<<<2048, 256, 0, stream>>>((const float*)d_in[0], (float*)d_out, wsW, wsP);
}
